// MultiheadSelfAttention_71829033058305
// MI455X (gfx1250) — matrix operand order repaired and checked by reading, not compiled, not run on hardware
//
#include <hip/hip_runtime.h>

// ---------------------------------------------------------------------------
// MI455X / gfx1250 multi-head self-attention (fwd), bf16 WMMA + fp32 accum.
// D=1024, H=16, Dh=64. Compute-bound -> maximize v_wmma_f32_16x16x32_bf16
// density. Softmax row reductions: fused DPP16 VOP2 ops (1 instr/step).
// ---------------------------------------------------------------------------

typedef __attribute__((ext_vector_type(16))) __bf16 bf16x16;
typedef __attribute__((ext_vector_type(8)))  __bf16 bf16x8;
typedef __attribute__((ext_vector_type(4)))  __bf16 bf16x4;
typedef __attribute__((ext_vector_type(8)))  float  f32x8;
typedef __attribute__((ext_vector_type(4)))  float  f32x4;

#define DMODEL 1024
#define NHEADS 16
#define HDIM   64

constexpr int EPI_ROPE = 0;  // RoPE (+qscale) + bf16 pack to [B,H,L,Dh]
constexpr int EPI_VT   = 1;  // bf16 pack transposed to [B,H,Dh,L]
constexpr int EPI_F32  = 2;  // plain fp32 [M,N] store

__device__ __forceinline__ bf16x16 cat8(bf16x8 lo, bf16x8 hi) {
    union { bf16x16 v; bf16x8 h[2]; } u;
    u.h[0] = lo; u.h[1] = hi;
    return u.v;
}

__device__ __forceinline__ f32x8 wmma_bf16(bf16x16 a, bf16x16 b, f32x8 c) {
    // (neg_a, A, neg_b, B, c_mod, C, reuse_a, reuse_b)
    return __builtin_amdgcn_wmma_f32_16x16x32_bf16(false, a, false, b,
                                                   (short)0, c, false, false);
}

// DPP16 helpers (wave32: one "row" = 16 lanes = one WMMA column group).
template<int CTRL>
__device__ __forceinline__ float mov_dpp_f(float x) {
    return __int_as_float(__builtin_amdgcn_mov_dpp(__float_as_int(x),
                                                   CTRL, 0xF, 0xF, true));
}

// Fused rotate-butterfly steps: dst = op(ror(src0), src1) in ONE VOP2+DPP16
// instruction. After ror 1,2,4,8 every lane holds the 16-lane row reduction.
#define ROR_MAX(v, n)                                                          \
    asm("v_max_num_f32_dpp %0, %1, %1 row_ror:" #n                             \
        " row_mask:0xf bank_mask:0xf bound_ctrl:1" : "=v"(v) : "v"(v))
#define ROR_ADD(v, n)                                                          \
    asm("v_add_f32_dpp %0, %1, %1 row_ror:" #n                                 \
        " row_mask:0xf bank_mask:0xf bound_ctrl:1" : "=v"(v) : "v"(v))

__device__ __forceinline__ float row16_max(float v) {
    ROR_MAX(v, 1); ROR_MAX(v, 2); ROR_MAX(v, 4); ROR_MAX(v, 8);
    return v;
}
__device__ __forceinline__ float row16_sum(float v) {
    ROR_ADD(v, 1); ROR_ADD(v, 2); ROR_ADD(v, 4); ROR_ADD(v, 8);
    return v;
}

// ---------------------------------------------------------------------------
__global__ void cvt_f32_bf16_v4(const f32x4* __restrict__ in,
                                bf16x4* __restrict__ out, int n4) {
    int i = blockIdx.x * blockDim.x + threadIdx.x;
    if (i < n4) out[i] = __builtin_convertvector(in[i], bf16x4);
}

// ---------------------------------------------------------------------------
// C[M,N] = A[M,K] @ W[N,K]^T, bf16 operands, fp32 accumulation.
// One wave computes a 32x64 output tile: 2 A operands x 4 B operands
// -> 8 WMMAs per K=32 step against 12x16B loads (good L2 operand reuse).
// Operand layouts (wave32, 16-bit WMMA):
//   A 16x32: lane m (m<16): K {0..7} in v0..3, {16..23} in v4..7;
//            lane m+16:     K {8..15},{24..31}  -> two 16B chunks per lane.
//   B 32x16: lane n holds W row n, same K split as A; lanes 0-15: K {0..7},{16..23},
//            lanes 16-31: K {8..15},{24..31} -> two 16B chunks per lane.
template<int EPI>
__global__ __launch_bounds__(256)
void gemm_bf16_k(const __bf16* __restrict__ A, const __bf16* __restrict__ W,
                 void* __restrict__ outp, const int* __restrict__ pos,
                 int M, int N, int K, int Bsz, int Lseq, float qscale)
{
    const int lane = threadIdx.x & 31;
    const int wave = threadIdx.x >> 5;
    const int ln   = lane & 15;
    const int sel  = lane >> 4;
    const int m0 = (blockIdx.y * 8 + wave) * 32;
    const int n0 = blockIdx.x * 64;

    f32x8 acc[2][4] = {};
    const __bf16* pa0 = A + (size_t)(m0 + ln) * K;
    const __bf16* pa1 = A + (size_t)(m0 + 16 + ln) * K;

    for (int k0 = 0; k0 < K; k0 += 32) {
        bf16x16 av0 = cat8(*(const bf16x8*)(pa0 + k0 + sel * 8),
                           *(const bf16x8*)(pa0 + k0 + 16 + sel * 8));
        bf16x16 av1 = cat8(*(const bf16x8*)(pa1 + k0 + sel * 8),
                           *(const bf16x8*)(pa1 + k0 + 16 + sel * 8));
#pragma unroll
        for (int t = 0; t < 4; ++t) {
            const __bf16* pb = W + (size_t)(n0 + t * 16 + ln) * K + k0;
            bf16x16 bv = cat8(*(const bf16x8*)(pb + sel * 8), *(const bf16x8*)(pb + 16 + sel * 8));
            acc[0][t] = wmma_bf16(av0, bv, acc[0][t]);
            acc[1][t] = wmma_bf16(av1, bv, acc[1][t]);
        }
    }

    // C layout: VGPR r, lanes 0-15 = row r, lanes 16-31 = row r+8; col = ln.
    if (EPI == EPI_ROPE) {
        // invf depends only on the column -> hoist out of the row loops.
        float invf[4];
        int   hh[4], dd[4];
#pragma unroll
        for (int t = 0; t < 4; ++t) {
            const int col = n0 + t * 16 + ln;
            hh[t] = col >> 6; dd[t] = col & 63;
            invf[t] = __expf(-(float)(dd[t] >> 1) * 0.28782313662425570f); // ln(1e4)/32
        }
#pragma unroll
        for (int mi = 0; mi < 2; ++mi)
#pragma unroll
        for (int r = 0; r < 8; ++r) {
            const int row = m0 + mi * 16 + r + sel * 8;
            const int b = row / Lseq, l = row - b * Lseq;
            const float p = (float)pos[l];
#pragma unroll
            for (int t = 0; t < 4; ++t) {
                float v = acc[mi][t][r];
                float partner = mov_dpp_f<0xB1>(v);       // lane^1 (quad_perm)
                float s, c; __sincosf(p * invf[t], &s, &c);
                float outv = (dd[t] & 1) ? (partner * s + v * c)
                                         : (v * c - partner * s);
                ((__bf16*)outp)[(((size_t)b * NHEADS + hh[t]) * Lseq + l) * HDIM + dd[t]]
                    = (__bf16)(outv * qscale);            // fold 1/sqrt(Dh) into Q
            }
        }
    } else {
#pragma unroll
        for (int mi = 0; mi < 2; ++mi)
#pragma unroll
        for (int t = 0; t < 4; ++t) {
            const int col = n0 + t * 16 + ln;
#pragma unroll
            for (int r = 0; r < 8; ++r) {
                const int row = m0 + mi * 16 + r + sel * 8;
                float v = acc[mi][t][r];
                if (EPI == EPI_F32) {
                    ((float*)outp)[(size_t)row * N + col] = v;
                } else { // EPI_VT
                    const int b = row / Lseq, l = row - b * Lseq;
                    const int h = col >> 6, d = col & 63;
                    ((__bf16*)outp)[(((size_t)b * NHEADS + h) * HDIM + d) * Lseq + l]
                        = (__bf16)v;
                }
            }
        }
    }
}

// ---------------------------------------------------------------------------
// One key-chunk step of the flash loop. NT = number of 16-key score tiles
// (2 -> 32 keys, 4 -> 64 keys). MASKED only for the single diagonal chunk.
// Kh: K base for (b,h) [L,Dh];  Vh: V^T base for (b,h) [Dh,L].
template<int NT, bool MASKED>
__device__ __forceinline__ void attn_chunk(
    const __bf16* __restrict__ Kh, const __bf16* __restrict__ Vh,
    int Lseq, int k0, int q0, int ln, int sel,
    bf16x16 aq0, bf16x16 aq1, __bf16* pp,
    f32x8 (&o)[4], float (&Mr)[8], float (&Ls)[8])
{
    // S = Q @ K^T : 2 WMMAs (Dh=64 in two K=32 steps) per 16-key tile
    f32x8 s[NT];
#pragma unroll
    for (int nt = 0; nt < NT; ++nt) {
        const __bf16* pk = Kh + (size_t)(k0 + nt * 16 + ln) * HDIM;
        bf16x16 b0 = cat8(*(const bf16x8*)(pk +      sel * 8), *(const bf16x8*)(pk + 16 + sel * 8));
        bf16x16 b1 = cat8(*(const bf16x8*)(pk + 32 + sel * 8), *(const bf16x8*)(pk + 48 + sel * 8));
        s[nt] = f32x8{};
        s[nt] = wmma_bf16(aq0, b0, s[nt]);
        s[nt] = wmma_bf16(aq1, b1, s[nt]);
    }
    if (MASKED) {
#pragma unroll
        for (int nt = 0; nt < NT; ++nt)
#pragma unroll
            for (int r = 0; r < 8; ++r) {
                const int key = k0 + nt * 16 + ln;
                const int qr  = q0 + r + sel * 8;
                if (key > qr) s[nt][r] = -1.0e9f;
            }
    }
    // online softmax: row reductions via fused DPP16 rotate-butterflies
#pragma unroll
    for (int r = 0; r < 8; ++r) {
        float mx = s[0][r];
#pragma unroll
        for (int nt = 1; nt < NT; ++nt) mx = fmaxf(mx, s[nt][r]);
        mx = row16_max(mx);
        const float mnew  = fmaxf(Mr[r], mx);
        const float alpha = __expf(Mr[r] - mnew);
        Mr[r] = mnew;
        float ps = 0.0f;
#pragma unroll
        for (int nt = 0; nt < NT; ++nt) {
            s[nt][r] = __expf(s[nt][r] - mnew);
            ps += s[nt][r];
        }
        ps = row16_sum(ps);
        Ls[r] = Ls[r] * alpha + ps;
#pragma unroll
        for (int t = 0; t < 4; ++t) o[t][r] *= alpha;
    }
    // P: C-layout -> LDS (bf16, row stride 64) -> A-layout. Per-wave region,
    // LDS ops are in-order per wave; just fence writes->reads.
#pragma unroll
    for (int nt = 0; nt < NT; ++nt)
#pragma unroll
        for (int r = 0; r < 8; ++r)
            pp[(r + sel * 8) * 64 + nt * 16 + ln] = (__bf16)s[nt][r];
    asm volatile("s_wait_dscnt 0" ::: "memory");
    // O += P @ V  (one K=32 step per 32 keys; Vt rows contiguous per lane)
#pragma unroll
    for (int kk = 0; kk < NT / 2; ++kk) {
        bf16x16 ap = cat8(*(const bf16x8*)(pp + ln * 64 + kk * 32 +      sel * 8),
                          *(const bf16x8*)(pp + ln * 64 + kk * 32 + 16 + sel * 8));
#pragma unroll
        for (int t = 0; t < 4; ++t) {
            const __bf16* pv = Vh + (size_t)(t * 16 + ln) * Lseq
                             + k0 + kk * 32;
            bf16x16 bv = cat8(*(const bf16x8*)(pv + sel * 8), *(const bf16x8*)(pv + 16 + sel * 8));
            o[t] = wmma_bf16(ap, bv, o[t]);
        }
    }
}

// ---------------------------------------------------------------------------
// Flash attention: one wave = 16 queries of one (b,h). Unmasked 64-key main
// loop, optional unmasked 32-key chunk, one masked diagonal chunk.
// Q:[B,H,L,Dh] (pre-scaled by 1/8) K:[B,H,L,Dh] Vt:[B,H,Dh,L]  (all bf16)
__global__ __launch_bounds__(256)
void flash_attn_k(const __bf16* __restrict__ Q, const __bf16* __restrict__ Kc,
                  const __bf16* __restrict__ Vt, __bf16* __restrict__ Y,
                  int Bsz, int Lseq)
{
    __shared__ __bf16 psh[8][16 * 64];   // per-wave P-transpose scratch (2KB each)
    const int lane = threadIdx.x & 31;
    const int wave = threadIdx.x >> 5;
    const int ln = lane & 15, sel = lane >> 4;
    const int ntile = Lseq >> 4;
    const int wid = blockIdx.x * 8 + wave;
    const int qt = wid % ntile;
    const int h  = (wid / ntile) % NHEADS;
    const int b  = wid / (ntile * NHEADS);
    const int q0 = qt * 16;
    const size_t bh = (size_t)b * NHEADS + h;

    const __bf16* Kh = Kc + bh * Lseq * HDIM;
    const __bf16* Vh = Vt + bh * HDIM * Lseq;

    // Q tile as two A operands (K = Dh split 0..31 / 32..63), kept in VGPRs
    const __bf16* pq = Q + (bh * Lseq + q0 + ln) * HDIM;
    bf16x16 aq0 = cat8(*(const bf16x8*)(pq +      sel * 8),
                       *(const bf16x8*)(pq + 16 + sel * 8));
    bf16x16 aq1 = cat8(*(const bf16x8*)(pq + 32 + sel * 8),
                       *(const bf16x8*)(pq + 48 + sel * 8));

    f32x8 o[4] = {};
    float Mr[8], Ls[8];
#pragma unroll
    for (int r = 0; r < 8; ++r) { Mr[r] = -3.0e38f; Ls[r] = 0.0f; }

    __bf16* pp = &psh[wave][0];

    const int tail = (q0 + 15) & ~31;   // start of the (single) masked chunk
    int k0 = 0;
    for (; k0 + 64 <= tail; k0 += 64)
        attn_chunk<4, false>(Kh, Vh, Lseq, k0, q0, ln, sel, aq0, aq1, pp, o, Mr, Ls);
    for (; k0 < tail; k0 += 32)
        attn_chunk<2, false>(Kh, Vh, Lseq, k0, q0, ln, sel, aq0, aq1, pp, o, Mr, Ls);
    attn_chunk<2, true>(Kh, Vh, Lseq, tail, q0, ln, sel, aq0, aq1, pp, o, Mr, Ls);

    // normalize and store to Y [B, L, H*Dh] bf16 (input to the Wo GEMM)
#pragma unroll
    for (int r = 0; r < 8; ++r) {
        const float inv = 1.0f / Ls[r];
        const int row = q0 + r + sel * 8;
#pragma unroll
        for (int t = 0; t < 4; ++t) {
            const int col = h * HDIM + t * 16 + ln;
            Y[((size_t)b * Lseq + row) * DMODEL + col] = (__bf16)(o[t][r] * inv);
        }
    }
}

// ---------------------------------------------------------------------------
extern "C" void kernel_launch(void* const* d_in, const int* in_sizes, int n_in,
                              void* d_out, int out_size, void* d_ws, size_t ws_size,
                              hipStream_t stream)
{
    const float* x  = (const float*)d_in[0];
    const float* wq = (const float*)d_in[1];
    const float* wk = (const float*)d_in[2];
    const float* wv = (const float*)d_in[3];
    const float* wo = (const float*)d_in[4];
    const int*  pos = (const int*)d_in[5];

    const int Lseq = in_sizes[5];                       // 2048
    const int Bsz  = in_sizes[0] / (Lseq * DMODEL);     // 4
    const int M    = Bsz * Lseq;                        // 8192

    // Workspace carve-up (bf16): x, 4 weights, q, k, vt, y  (~88 MB total)
    char* w = (char*)d_ws;
    __bf16* xb  = (__bf16*)w; w += (size_t)M * DMODEL * sizeof(__bf16);
    __bf16* wqb = (__bf16*)w; w += (size_t)DMODEL * DMODEL * sizeof(__bf16);
    __bf16* wkb = (__bf16*)w; w += (size_t)DMODEL * DMODEL * sizeof(__bf16);
    __bf16* wvb = (__bf16*)w; w += (size_t)DMODEL * DMODEL * sizeof(__bf16);
    __bf16* wob = (__bf16*)w; w += (size_t)DMODEL * DMODEL * sizeof(__bf16);
    __bf16* qb  = (__bf16*)w; w += (size_t)M * DMODEL * sizeof(__bf16);
    __bf16* kb  = (__bf16*)w; w += (size_t)M * DMODEL * sizeof(__bf16);
    __bf16* vt  = (__bf16*)w; w += (size_t)M * DMODEL * sizeof(__bf16);
    __bf16* yb  = (__bf16*)w; w += (size_t)M * DMODEL * sizeof(__bf16);

    const int n1 = (M * DMODEL) / 4, n2 = (DMODEL * DMODEL) / 4;
    cvt_f32_bf16_v4<<<(n1 + 255) / 256, 256, 0, stream>>>((const f32x4*)x,  (bf16x4*)xb,  n1);
    cvt_f32_bf16_v4<<<(n2 + 255) / 256, 256, 0, stream>>>((const f32x4*)wq, (bf16x4*)wqb, n2);
    cvt_f32_bf16_v4<<<(n2 + 255) / 256, 256, 0, stream>>>((const f32x4*)wk, (bf16x4*)wkb, n2);
    cvt_f32_bf16_v4<<<(n2 + 255) / 256, 256, 0, stream>>>((const f32x4*)wv, (bf16x4*)wvb, n2);
    cvt_f32_bf16_v4<<<(n2 + 255) / 256, 256, 0, stream>>>((const f32x4*)wo, (bf16x4*)wob, n2);

    dim3 ggrid(DMODEL / 64, M / 256);   // 8 waves/block, 32x64 tile per wave
    gemm_bf16_k<EPI_ROPE><<<ggrid, 256, 0, stream>>>(xb, wqb, qb, pos,
                                 M, DMODEL, DMODEL, Bsz, Lseq, 0.125f);
    gemm_bf16_k<EPI_ROPE><<<ggrid, 256, 0, stream>>>(xb, wkb, kb, pos,
                                 M, DMODEL, DMODEL, Bsz, Lseq, 1.0f);
    gemm_bf16_k<EPI_VT><<<ggrid, 256, 0, stream>>>(xb, wvb, vt, nullptr,
                                 M, DMODEL, DMODEL, Bsz, Lseq, 1.0f);

    const int nwaves = Bsz * NHEADS * (Lseq / 16);      // 8192
    flash_attn_k<<<nwaves / 8, 256, 0, stream>>>(qb, kb, vt, yb, Bsz, Lseq);

    gemm_bf16_k<EPI_F32><<<ggrid, 256, 0, stream>>>(yb, wob, d_out, nullptr,
                                 M, DMODEL, DMODEL, Bsz, Lseq, 1.0f);
}
